// RepresentationHead_76072460747049
// MI455X (gfx1250) — matrix operand order repaired and checked by reading, not compiled, not run on hardware
//
#include <hip/hip_runtime.h>
#include <math.h>

typedef _Float16 v16h __attribute__((ext_vector_type(16)));
typedef _Float16 v8h  __attribute__((ext_vector_type(8)));
typedef float    v8f  __attribute__((ext_vector_type(8)));

#define DD 256    // feature dim D
#define HH 512    // hidden H
#define SS 8192   // slots S
#define RR 256    // rep dim R
#define LSTR 264  // LDS row stride in halves: 528B = 33 x 16B (odd) -> conflict-free b128
#define MROWS 64  // slots per block
#define MT 4      // 16-row M-tiles per block

// Branchless exact-GELU: erf via Abramowitz-Stegun 7.1.26 (|err| <= 1.5e-7).
// No EXEC divergence -> co-executes cleanly under the XDL WMMA pipe.
__device__ __forceinline__ float gelu_exact(float x) {
    float z = fabsf(x) * 0.70710678118654752f;
    float t = __builtin_amdgcn_rcpf(1.0f + 0.3275911f * z);
    float p = t * (0.254829592f + t * (-0.284496736f + t * (1.421413741f +
              t * (-1.453152027f + t * 1.061405429f))));
    float erfv = 1.0f - p * __expf(-z * z);
    erfv = copysignf(erfv, x);
    return 0.5f * x * (1.0f + erfv);
}

// ---------- kernel 0: coalesced LDS-tiled transpose+convert to f16 [N][K] ----------
// dw1/rw1: [256][512] f32 -> [512][256] f16 ; vw: [256][256] f32 -> [256][256] f16
__global__ __launch_bounds__(256) void k_cvt(const float* __restrict__ dw1,
                                             const float* __restrict__ rw1,
                                             const float* __restrict__ vw,
                                             _Float16* __restrict__ dw1t,
                                             _Float16* __restrict__ rw1t,
                                             _Float16* __restrict__ vwt) {
    __shared__ float tbuf[32][33];
    int bid = blockIdx.x;
    const float* src;
    _Float16* dst;
    int N;
    if (bid < 128)      { src = dw1; dst = dw1t; N = HH; }
    else if (bid < 256) { src = rw1; dst = rw1t; N = HH; bid -= 128; }
    else                { src = vw;  dst = vwt;  N = RR; bid -= 256; }
    int ntn = N / 32;
    int k0 = (bid / ntn) * 32, n0 = (bid % ntn) * 32;
    int tx = threadIdx.x & 31, ty = threadIdx.x >> 5;  // ty: 0..7
#pragma unroll
    for (int j = 0; j < 4; ++j) {
        int k = ty + j * 8;
        tbuf[k][tx] = src[(size_t)(k0 + k) * N + n0 + tx];
    }
    __syncthreads();
#pragma unroll
    for (int j = 0; j < 4; ++j) {
        int n = ty + j * 8;
        dst[(size_t)(n0 + n) * DD + k0 + tx] = (_Float16)tbuf[tx][n];
    }
}

// ---------- shared helper: LayerNorm 32 rows (256 threads, 8/row, 32 elems/thr) ----------
__device__ __forceinline__ void ln32(const float* __restrict__ x, long tileBase,
                                     const float* __restrict__ g,
                                     const float* __restrict__ be,
                                     _Float16* nrm, int row0) {
    int tid = threadIdx.x;
    int row = row0 + (tid >> 3), lane8 = tid & 7;
    const float* xr = x + tileBase + (long)row * DD + lane8 * 32;
    float4 v[8];
#pragma unroll
    for (int i = 0; i < 8; ++i) v[i] = ((const float4*)xr)[i];
    float s = 0.f, s2 = 0.f;
#pragma unroll
    for (int i = 0; i < 8; ++i) {
        s  += v[i].x + v[i].y + v[i].z + v[i].w;
        s2 += v[i].x * v[i].x + v[i].y * v[i].y + v[i].z * v[i].z + v[i].w * v[i].w;
    }
#pragma unroll
    for (int o = 1; o < 8; o <<= 1) { s += __shfl_xor(s, o, 8); s2 += __shfl_xor(s2, o, 8); }
    float mu  = s * (1.0f / DD);
    float var = fmaxf(s2 * (1.0f / DD) - mu * mu, 0.0f);
    float inv = rsqrtf(var + 1e-5f);
    _Float16* dst = nrm + row * LSTR + lane8 * 32;
    const float* gg = g + lane8 * 32;
    const float* bb = be + lane8 * 32;
#pragma unroll
    for (int i = 0; i < 8; ++i) {
        dst[i * 4 + 0] = (_Float16)((v[i].x - mu) * inv * gg[i * 4 + 0] + bb[i * 4 + 0]);
        dst[i * 4 + 1] = (_Float16)((v[i].y - mu) * inv * gg[i * 4 + 1] + bb[i * 4 + 1]);
        dst[i * 4 + 2] = (_Float16)((v[i].z - mu) * inv * gg[i * 4 + 2] + bb[i * 4 + 2]);
        dst[i * 4 + 3] = (_Float16)((v[i].w - mu) * inv * gg[i * 4 + 3] + bb[i * 4 + 3]);
    }
}

// A-fragment loader for v_wmma_f32_16x16x32_f16 (one 16x32 k-step from LDS).
// Lanes 0-15 row M=lane hold K {0..15}; lanes 16-31 hold K {16..31} (matches B).
__device__ __forceinline__ v16h load_a(const _Float16* rbase, int k, int akoff) {
    v8h lo = *(const v8h*)(rbase + k * 32 + 2 * akoff);
    v8h hi = *(const v8h*)(rbase + k * 32 + 2 * akoff + 8);
    return __builtin_shufflevector(lo, hi, 0, 1, 2, 3, 4, 5, 6, 7,
                                   8, 9, 10, 11, 12, 13, 14, 15);
}

// ---------- kernel 1: layernorm + drive/resistance MLPs -> logits ----------
// 64 slots/block, 8 waves. Waves 0-3: drive (H=512 -> 8 N-tiles each).
// Waves 4-7: resistance. B-fragments register-stationary across 4 M-tiles.
__global__ __launch_bounds__(256) void k_logits(const float* __restrict__ x,
                                                const float* __restrict__ g,
                                                const float* __restrict__ be,
                                                const _Float16* __restrict__ dw1t,
                                                const float* __restrict__ db1,
                                                const float* __restrict__ dw2,
                                                const float* __restrict__ db2,
                                                const _Float16* __restrict__ rw1t,
                                                const float* __restrict__ rb1,
                                                const float* __restrict__ rw2,
                                                const float* __restrict__ rb2,
                                                float* __restrict__ logits) {
    __shared__ _Float16 nrm[MROWS * LSTR];
    __shared__ float accD[MROWS], accR[MROWS];
    long tileBase = (long)blockIdx.x * MROWS * DD;
    int tid = threadIdx.x;
    ln32(x, tileBase, g, be, nrm, 0);
    ln32(x, tileBase, g, be, nrm, 32);
    if (tid < MROWS) { accD[tid] = db2[0]; accR[tid] = rb2[0]; }
    __syncthreads();

    int w = tid >> 5, l = tid & 31;
    int mrow = l & 15;
    int akoff = (l < 16) ? 0 : 8;
    int ncolBase = l & 15;
    int kboff = (l < 16) ? 0 : 16;
    bool isDrive = (w < 4);
    const _Float16* wt = isDrive ? dw1t : rw1t;
    const float* b1p = isDrive ? db1 : rb1;
    const float* w2p = isDrive ? dw2 : rw2;
    int tbase = (w & 3) * 8;

    float part[MT][8];
#pragma unroll
    for (int m = 0; m < MT; ++m)
#pragma unroll
        for (int r = 0; r < 8; ++r) part[m][r] = 0.f;

    for (int t = 0; t < 8; ++t) {
        int n = (tbase + t) * 16 + ncolBase;
        const _Float16* bp = wt + (size_t)n * DD + kboff;
        __builtin_prefetch(bp + 16 * DD, 0, 1);
        v16h Bf[8];
#pragma unroll
        for (int k = 0; k < 8; ++k) Bf[k] = *(const v16h*)(bp + k * 32);
        float b1 = b1p[n], w2 = w2p[n];
#pragma unroll
        for (int m = 0; m < MT; ++m) {
            const _Float16* rbase = nrm + (m * 16 + mrow) * LSTR;
            v8f c = {};
#pragma unroll
            for (int k = 0; k < 8; ++k) {
                v16h a = load_a(rbase, k, akoff);
                c = __builtin_amdgcn_wmma_f32_16x16x32_f16(false, a, false, Bf[k],
                                                           (short)0, c, false, false);
            }
#pragma unroll
            for (int r = 0; r < 8; ++r) part[m][r] += gelu_exact(c[r] + b1) * w2;
        }
    }

    float* acc = isDrive ? accD : accR;
#pragma unroll
    for (int m = 0; m < MT; ++m)
#pragma unroll
        for (int r = 0; r < 8; ++r) {
            float pv = part[m][r];
#pragma unroll
            for (int o = 1; o < 16; o <<= 1) pv += __shfl_xor(pv, o, 16);
            if ((l & 15) == 0)
                atomicAdd(&acc[m * 16 + r + ((l < 16) ? 0 : 8)], pv);
        }
    __syncthreads();
    if (tid < MROWS) {
        float dr = accD[tid], rs = accR[tid];
        float sp = (rs > 20.f) ? rs : log1pf(expf(rs));
        float sc = dr - sp;                       // WARMUP_ETA=1, EML_BIAS=0
        sc = fminf(fmaxf(sc, -3.0f), 3.0f);
        logits[(long)blockIdx.x * MROWS + tid] = sc;
    }
}

// ---------- kernel 2: masked softmax + top3 + rep zero-init ----------
__global__ __launch_bounds__(256) void k_softmax(const float* __restrict__ logits,
                                                 const int* __restrict__ mask,
                                                 float* __restrict__ weights,
                                                 float* __restrict__ rep,
                                                 float* __restrict__ topw,
                                                 float* __restrict__ topi) {
    int b = blockIdx.x, t = threadIdx.x;
    const float* lg = logits + (size_t)b * SS;
    const int* mk = mask + (size_t)b * SS;
    __shared__ float red[256];
    __shared__ int redi[256];
    rep[b * RR + t] = 0.0f;

    float mx = -INFINITY;
    for (int s = t; s < SS; s += 256) if (mk[s]) mx = fmaxf(mx, lg[s]);
    red[t] = mx; __syncthreads();
    for (int o = 128; o > 0; o >>= 1) { if (t < o) red[t] = fmaxf(red[t], red[t + o]); __syncthreads(); }
    mx = red[0]; __syncthreads();

    float sum = 0.f;
    for (int s = t; s < SS; s += 256) if (mk[s]) sum += expf(lg[s] - mx);
    red[t] = sum; __syncthreads();
    for (int o = 128; o > 0; o >>= 1) { if (t < o) red[t] += red[t + o]; __syncthreads(); }
    sum = red[0]; __syncthreads();
    float inv = (sum > 0.f) ? 1.0f / sum : 0.0f;

    float* wout = weights + (size_t)b * SS;
    for (int s = t; s < SS; s += 256)
        wout[s] = mk[s] ? expf(lg[s] - mx) * inv : 0.0f;

    int ex0 = -1, ex1 = -1;
    for (int j = 0; j < 3; ++j) {
        float bv = -1.0f; int bi = SS;
        for (int s = t; s < SS; s += 256) {
            if (s == ex0 || s == ex1) continue;
            float wv = mk[s] ? expf(lg[s] - mx) * inv : 0.0f;
            if (wv > bv || (wv == bv && s < bi)) { bv = wv; bi = s; }
        }
        red[t] = bv; redi[t] = bi; __syncthreads();
        for (int o = 128; o > 0; o >>= 1) {
            if (t < o) {
                if (red[t + o] > red[t] || (red[t + o] == red[t] && redi[t + o] < redi[t])) {
                    red[t] = red[t + o]; redi[t] = redi[t + o];
                }
            }
            __syncthreads();
        }
        if (t == 0) { topw[b * 3 + j] = red[0]; topi[b * 3 + j] = (float)redi[0]; }
        int win = redi[0];
        __syncthreads();
        if (j == 0) ex0 = win; else if (j == 1) ex1 = win;
    }
}

// ---------- kernel 3: recompute layernorm, values GEMM (+vb), weighted sum ----------
__global__ __launch_bounds__(256) void k_repr(const float* __restrict__ x,
                                              const float* __restrict__ g,
                                              const float* __restrict__ be,
                                              const _Float16* __restrict__ vwt,
                                              const float* __restrict__ vb,
                                              const float* __restrict__ weights,
                                              float* __restrict__ rep) {
    __shared__ _Float16 nrm[MROWS * LSTR];
    __shared__ float wslot[MROWS];
    int tid = threadIdx.x;
    int b = blockIdx.x / (SS / MROWS);  // 128 blocks per batch
    long tileBase = (long)blockIdx.x * MROWS * DD;
    ln32(x, tileBase, g, be, nrm, 0);
    ln32(x, tileBase, g, be, nrm, 32);
    if (tid < MROWS) wslot[tid] = weights[(long)blockIdx.x * MROWS + tid];
    __syncthreads();

    int w = tid >> 5, l = tid & 31;
    int mrow = l & 15;
    int akoff = (l < 16) ? 0 : 8;
    int ncolBase = l & 15;
    int kboff = (l < 16) ? 0 : 16;

    // 16 N-tiles over R=256: 2 per wave; B register-stationary across 4 M-tiles
#pragma unroll
    for (int j = 0; j < 2; ++j) {
        int n = (w * 2 + j) * 16 + ncolBase;
        const _Float16* bp = vwt + (size_t)n * DD + kboff;
        v16h Bf[8];
#pragma unroll
        for (int k = 0; k < 8; ++k) Bf[k] = *(const v16h*)(bp + k * 32);
        float bv = vb[n];
#pragma unroll
        for (int m = 0; m < MT; ++m) {
            const _Float16* rbase = nrm + (m * 16 + mrow) * LSTR;
            v8f c = {};
#pragma unroll
            for (int k = 0; k < 8; ++k) {
                v16h a = load_a(rbase, k, akoff);
                c = __builtin_amdgcn_wmma_f32_16x16x32_f16(false, a, false, Bf[k],
                                                           (short)0, c, false, false);
            }
            float colsum = 0.f;
#pragma unroll
            for (int r = 0; r < 8; ++r) {
                int mm = m * 16 + r + ((l < 16) ? 0 : 8);
                colsum += (c[r] + bv) * wslot[mm];
            }
            colsum += __shfl_xor(colsum, 16, 32);  // combine the two M-halves
            if (l < 16) atomicAdd(&rep[b * RR + n], colsum);
        }
    }
}

extern "C" void kernel_launch(void* const* d_in, const int* in_sizes, int n_in,
                              void* d_out, int out_size, void* d_ws, size_t ws_size,
                              hipStream_t stream) {
    (void)in_sizes; (void)n_in; (void)out_size; (void)ws_size;
    const float* slot = (const float*)d_in[0];
    const int*   mask = (const int*)d_in[1];
    const float* g    = (const float*)d_in[2];
    const float* be   = (const float*)d_in[3];
    const float* dw1  = (const float*)d_in[4];
    const float* db1  = (const float*)d_in[5];
    const float* dw2  = (const float*)d_in[6];
    const float* db2  = (const float*)d_in[7];
    const float* rw1  = (const float*)d_in[8];
    const float* rb1  = (const float*)d_in[9];
    const float* rw2  = (const float*)d_in[10];
    const float* rb2  = (const float*)d_in[11];
    const float* vw   = (const float*)d_in[12];
    const float* vb   = (const float*)d_in[13];

    float* out = (float*)d_out;
    // tuple layout: representation[8192], weights[262144], top_indices[96],
    //               top_weights[96], logits[262144]
    float* rep = out;
    float* wts = out + 32 * 256;
    float* ti  = out + 32 * 256 + 32 * 8192;
    float* tw  = ti + 96;
    float* lgt = tw + 96;

    _Float16* dw1t = (_Float16*)d_ws;
    _Float16* rw1t = dw1t + (size_t)HH * DD;
    _Float16* vwt  = rw1t + (size_t)HH * DD;

    const int tiles = (32 * SS) / MROWS;  // 4096
    k_cvt<<<320, 256, 0, stream>>>(dw1, rw1, vw, dw1t, rw1t, vwt);
    k_logits<<<tiles, 256, 0, stream>>>(slot, g, be, dw1t, db1, dw2, db2,
                                        rw1t, rb1, rw2, rb2, lgt);
    k_softmax<<<32, 256, 0, stream>>>(lgt, mask, wts, rep, tw, ti);
    k_repr<<<tiles, 256, 0, stream>>>(slot, g, be, vwt, vb, wts, rep);
}
